// TopKGate_84705345012182
// MI455X (gfx1250) — matrix operand order repaired and checked by reading, not compiled, not run on hardware
//
#include <hip/hip_runtime.h>
#include <hip/hip_bf16.h>

#define NUM_TOKENS 8192
#define MODEL_DIM  2048
#define NUM_EXPERTS 64
#define CAPACITY   256   // ceil(2*8192/64*1.0)

typedef __attribute__((ext_vector_type(16))) __bf16 v16bf;
typedef __attribute__((ext_vector_type(8)))  float  v8f;

// ---------------------------------------------------------------------------
// Kernel 0: transpose gate weight [E,K] fp32 -> [K,E] bf16 so the WMMA
// B-fragment (lane&15 = N, elements = K) is a contiguous 32B load across 16 lanes.
// ---------------------------------------------------------------------------
__global__ __launch_bounds__(256)
void TopKGate_prep_wgT(const float* __restrict__ wg, __hip_bfloat16* __restrict__ wgT) {
    int i = blockIdx.x * 256 + threadIdx.x;        // i indexes [k][n], k = i>>6, n = i&63
    int n = i & (NUM_EXPERTS - 1);
    int k = i >> 6;
    wgT[i] = __float2bfloat16(wg[(size_t)n * MODEL_DIM + k]);
}

// ---------------------------------------------------------------------------
// Kernel 1: logits = x @ wg^T via v_wmma_f32_16x16x32_bf16.
// One wave -> 16 tokens x 64 experts (4 accumulator tiles), K-loop of 32.
// A layout (16-bit A 16x32): lane&15 = M, lane>=16 -> K-half 8;
//   elems 0..7 = K+0..7, elems 8..15 = K+16..23.
// B layout (16-bit B 32x16): lane&15 = N, elems = K in the same order as A.
// C/D layout: lane&15 = N, VGPR idx = M (lanes>=16: M+8).
// ---------------------------------------------------------------------------
__global__ __launch_bounds__(256)
void TopKGate_gemm(const float* __restrict__ x,
                   const __hip_bfloat16* __restrict__ wgT,
                   float* __restrict__ logits) {
    const int lane   = threadIdx.x & 31;
    const int waveId = (blockIdx.x << 3) | (threadIdx.x >> 5);
    const int m0     = waveId << 4;                  // 16 tokens per wave
    const int rowA   = lane & 15;
    const int kOff   = (lane >> 4) << 3;             // 0 or 8

    const float* aRow = x + (size_t)(m0 + rowA) * MODEL_DIM + kOff;

    v8f acc0 = {}, acc1 = {}, acc2 = {}, acc3 = {};

    for (int k = 0; k < MODEL_DIM; k += 32) {
        __builtin_prefetch(aRow + k + 64, 0, 1);     // global_prefetch next chunks
        const float4* ap = reinterpret_cast<const float4*>(aRow + k);
        float4 p0 = ap[0], p1 = ap[1];               // K = kOff+0..7
        float4 p2 = ap[4], p3 = ap[5];               // K = kOff+16..23
        v16bf a;
        a[0]  = (__bf16)p0.x; a[1]  = (__bf16)p0.y; a[2]  = (__bf16)p0.z; a[3]  = (__bf16)p0.w;
        a[4]  = (__bf16)p1.x; a[5]  = (__bf16)p1.y; a[6]  = (__bf16)p1.z; a[7]  = (__bf16)p1.w;
        a[8]  = (__bf16)p2.x; a[9]  = (__bf16)p2.y; a[10] = (__bf16)p2.z; a[11] = (__bf16)p2.w;
        a[12] = (__bf16)p3.x; a[13] = (__bf16)p3.y; a[14] = (__bf16)p3.z; a[15] = (__bf16)p3.w;

        // lane&15 = N column of wgT; elems = K in same order as A (kOff+0..7, kOff+16..23)
        const __hip_bfloat16* bp = wgT + (size_t)(k + kOff) * NUM_EXPERTS + rowA;
        v16bf b0, b1, b2, b3;
#pragma unroll
        for (int s = 0; s < 16; ++s) {
            const __hip_bfloat16* q = bp + (size_t)(s < 8 ? s : s + 8) * NUM_EXPERTS;
            b0[s] = __builtin_bit_cast(__bf16, q[0]);
            b1[s] = __builtin_bit_cast(__bf16, q[16]);
            b2[s] = __builtin_bit_cast(__bf16, q[32]);
            b3[s] = __builtin_bit_cast(__bf16, q[48]);
        }

        acc0 = __builtin_amdgcn_wmma_f32_16x16x32_bf16(false, a, false, b0, (short)0, acc0, false, false);
        acc1 = __builtin_amdgcn_wmma_f32_16x16x32_bf16(false, a, false, b1, (short)0, acc1, false, false);
        acc2 = __builtin_amdgcn_wmma_f32_16x16x32_bf16(false, a, false, b2, (short)0, acc2, false, false);
        acc3 = __builtin_amdgcn_wmma_f32_16x16x32_bf16(false, a, false, b3, (short)0, acc3, false, false);
    }

    const int mBase = m0 + ((lane >> 4) << 3);
    const int n     = lane & 15;
    float* outRow = logits + (size_t)mBase * NUM_EXPERTS + n;
#pragma unroll
    for (int i = 0; i < 8; ++i) {
        outRow[(size_t)i * NUM_EXPERTS +  0] = acc0[i];
        outRow[(size_t)i * NUM_EXPERTS + 16] = acc1[i];
        outRow[(size_t)i * NUM_EXPERTS + 32] = acc2[i];
        outRow[(size_t)i * NUM_EXPERTS + 48] = acc3[i];
    }
}

// ---------------------------------------------------------------------------
// Gating: one wave32 per token, 2 experts per lane. Softmax + top1 + gumbel
// top2 + aux-loss partial sums (LDS ds_add_f32, then one global atomic/block).
// ---------------------------------------------------------------------------
__device__ __forceinline__ float gumbel_noise(unsigned h) {
    h ^= 2654435769u + 42u;                          // salt (stand-in for jax key 42)
    h ^= h >> 16; h *= 0x7feb352du;
    h ^= h >> 15; h *= 0x846ca68bu;
    h ^= h >> 16;
    float u = (float)(h >> 8) * (1.0f / 16777216.0f);
    u = fmaxf(u, 1.0e-12f);
    u = fminf(u, 0.99999994f);
    return -__logf(-__logf(u));
}

__global__ __launch_bounds__(256)
void TopKGate_gate(const float* __restrict__ logits,
                   int* __restrict__ e1, int* __restrict__ e2,
                   float* __restrict__ g1, float* __restrict__ g2,
                   float* __restrict__ me_sum, float* __restrict__ ce_sum) {
    __shared__ float sme[NUM_EXPERTS];
    __shared__ float sce[NUM_EXPERTS];
    if (threadIdx.x < NUM_EXPERTS) { sme[threadIdx.x] = 0.0f; sce[threadIdx.x] = 0.0f; }
    __syncthreads();

    const int lane = threadIdx.x & 31;
    const int t    = (blockIdx.x << 3) | (threadIdx.x >> 5);

    const float* row = logits + (size_t)t * NUM_EXPERTS;
    float la = row[lane], lb = row[lane + 32];

    float mx = fmaxf(la, lb);
#pragma unroll
    for (int off = 16; off; off >>= 1) mx = fmaxf(mx, __shfl_xor(mx, off, 32));
    float ea = __expf(la - mx), eb = __expf(lb - mx);
    float s = ea + eb;
#pragma unroll
    for (int off = 16; off; off >>= 1) s += __shfl_xor(s, off, 32);
    float inv = 1.0f / s;
    float ga = ea * inv, gb = eb * inv;

    // top-1 (first-occurrence tie-break like jnp.argmax)
    float v = la; int idx = lane;
    if (lb > v) { v = lb; idx = lane + 32; }
#pragma unroll
    for (int off = 16; off; off >>= 1) {
        float ov = __shfl_xor(v, off, 32);
        int   oi = __shfl_xor(idx, off, 32);
        if (ov > v || (ov == v && oi < idx)) { v = ov; idx = oi; }
    }
    const int idx1 = idx;

    // top-2 on gumbel-perturbed logits with expert-1 masked out
    float na = la + gumbel_noise((unsigned)(t * NUM_EXPERTS + lane));
    float nb = lb + gumbel_noise((unsigned)(t * NUM_EXPERTS + lane + 32));
    if (lane == idx1)      na = -3.0e38f;
    if (lane + 32 == idx1) nb = -3.0e38f;
    v = na; idx = lane;
    if (nb > v) { v = nb; idx = lane + 32; }
#pragma unroll
    for (int off = 16; off; off >>= 1) {
        float ov = __shfl_xor(v, off, 32);
        int   oi = __shfl_xor(idx, off, 32);
        if (ov > v || (ov == v && oi < idx)) { v = ov; idx = oi; }
    }
    const int idx2 = idx;

    float sel1 = (idx1 < 32) ? ga : gb;              // idx1/idx2 are wave-uniform
    float sel2 = (idx2 < 32) ? ga : gb;
    float g1v = __shfl(sel1, idx1 & 31, 32);
    float g2v = __shfl(sel2, idx2 & 31, 32);

    atomicAdd(&sme[lane], ga);
    atomicAdd(&sme[lane + 32], gb);
    if (lane == 0) {
        atomicAdd(&sce[idx1], 1.0f);
        e1[t] = idx1; e2[t] = idx2; g1[t] = g1v; g2[t] = g2v;
    }
    __syncthreads();
    if (threadIdx.x < NUM_EXPERTS) {
        atomicAdd(&me_sum[threadIdx.x], sme[threadIdx.x]);
        atomicAdd(&ce_sum[threadIdx.x], sce[threadIdx.x]);
    }
}

// ---------------------------------------------------------------------------
// Token-ordered per-expert prefix counts (the cumsum is order dependent, so
// this stays deterministic: one thread per expert scans tokens in order).
// locations2 gets +count(top1 over ALL tokens) per the reference.
// ---------------------------------------------------------------------------
__global__ void TopKGate_scan(const int* __restrict__ e1, const int* __restrict__ e2,
                              int* __restrict__ loc1, int* __restrict__ loc2) {
    const int e = threadIdx.x;                       // 64 threads
    int c = 0;
    for (int t = 0; t < NUM_TOKENS; ++t)
        if (e1[t] == e) loc1[t] = c++;
    const int base = c;                              // sum(mask1) for this expert
    int c2 = 0;
    for (int t = 0; t < NUM_TOKENS; ++t)
        if (e2[t] == e) loc2[t] = base + c2++;
}

// ---------------------------------------------------------------------------
// Sparse scatter into the (memset-zeroed) dense outputs: 2 nonzeros per token.
// ---------------------------------------------------------------------------
__global__ __launch_bounds__(256)
void TopKGate_scatter(const int* __restrict__ e1, const int* __restrict__ e2,
                      const int* __restrict__ loc1, const int* __restrict__ loc2,
                      const float* __restrict__ g1, const float* __restrict__ g2,
                      float* __restrict__ out) {
    const int t = blockIdx.x * 256 + threadIdx.x;
    const int i1 = e1[t], i2 = e2[t];
    const int l1 = loc1[t], l2 = loc2[t];
    const bool k1 = l1 < CAPACITY, k2 = l2 < CAPACITY;
    float a = k1 ? g1[t] : 0.0f;
    float b = k2 ? g2[t] : 0.0f;
    const float denom = fmaxf(a + b, 1.1920929e-07f);
    a /= denom; b /= denom;

    float* cw = out + 1;                                              // combine_weights
    float* dm = out + 1 + (size_t)NUM_TOKENS * NUM_EXPERTS * CAPACITY; // dispatch_mask
    const size_t base = (size_t)t * NUM_EXPERTS * CAPACITY;
    if (k1 && a != 0.0f) { size_t o = base + (size_t)i1 * CAPACITY + l1; cw[o] = a; dm[o] = 1.0f; }
    if (k2 && b != 0.0f) { size_t o = base + (size_t)i2 * CAPACITY + l2; cw[o] = b; dm[o] = 1.0f; }
}

__global__ void TopKGate_laux(const float* __restrict__ me_sum,
                              const float* __restrict__ ce_sum,
                              float* __restrict__ out) {
    if (threadIdx.x == 0) {
        float s = 0.0f;
        for (int e = 0; e < NUM_EXPERTS; ++e)
            s += (me_sum[e] * (1.0f / NUM_TOKENS)) * (ce_sum[e] * (1.0f / NUM_TOKENS));
        out[0] = (s * (1.0f / NUM_EXPERTS)) * (float)(NUM_EXPERTS * NUM_EXPERTS);
    }
}

// ---------------------------------------------------------------------------
extern "C" void kernel_launch(void* const* d_in, const int* in_sizes, int n_in,
                              void* d_out, int out_size, void* d_ws, size_t ws_size,
                              hipStream_t stream) {
    (void)in_sizes; (void)n_in; (void)ws_size;
    const float* x  = (const float*)d_in[0];
    const float* wg = (const float*)d_in[1];
    float* out = (float*)d_out;

    // workspace layout (~2.6 MB)
    char* wsb = (char*)d_ws;
    float*          logits = (float*)wsb;                                    // 8192*64*4 = 2 MB
    __hip_bfloat16* wgT    = (__hip_bfloat16*)(wsb + 2097152);               // 2048*64*2 = 256 KB
    int*   e1   = (int*)(wsb + 2097152 + 262144);                            // 32 KB each
    int*   e2   = e1 + NUM_TOKENS;
    int*   loc1 = e2 + NUM_TOKENS;
    int*   loc2 = loc1 + NUM_TOKENS;
    float* g1   = (float*)(loc2 + NUM_TOKENS);
    float* g2   = g1 + NUM_TOKENS;
    float* me_sum = g2 + NUM_TOKENS;                                         // 64 floats
    float* ce_sum = me_sum + NUM_EXPERTS;                                    // 64 floats

    // zero the sparse outputs (dominant cost: ~1 GB streaming memset) + accumulators
    hipMemsetAsync(d_out, 0, (size_t)out_size * sizeof(float), stream);
    hipMemsetAsync(me_sum, 0, 2 * NUM_EXPERTS * sizeof(float), stream);

    TopKGate_prep_wgT<<<(MODEL_DIM * NUM_EXPERTS) / 256, 256, 0, stream>>>(wg, wgT);
    TopKGate_gemm<<<NUM_TOKENS / 16 / 8, 256, 0, stream>>>(x, wgT, logits);
    TopKGate_gate<<<NUM_TOKENS / 8, 256, 0, stream>>>(logits, e1, e2, g1, g2, me_sum, ce_sum);
    TopKGate_scan<<<1, NUM_EXPERTS, 0, stream>>>(e1, e2, loc1, loc2);
    TopKGate_scatter<<<NUM_TOKENS / 256, 256, 0, stream>>>(e1, e2, loc1, loc2, g1, g2, out);
    TopKGate_laux<<<1, 32, 0, stream>>>(me_sum, ce_sum, out);
}
